// MoE_19980187861463
// MI455X (gfx1250) — matrix operand order repaired and checked by reading, not compiled, not run on hardware
//
#include <hip/hip_runtime.h>

// Problem constants (match reference)
#define N_TOK 16384
#define H_DIM 2048
#define F_DIM 2048
#define N_EXP 8
#define LN_EPS 1e-5f

typedef __bf16 bf16_t;
typedef __attribute__((ext_vector_type(16))) __bf16 v16bf;
typedef __attribute__((ext_vector_type(8)))  __bf16 v8bf;
typedef __attribute__((ext_vector_type(8)))  float  v8f;
typedef __attribute__((ext_vector_type(4)))  float  v4f;
typedef __attribute__((ext_vector_type(4)))  unsigned int u32x4;
typedef __attribute__((ext_vector_type(8)))  int   i32x8;
typedef __attribute__((ext_vector_type(4)))  int   i32x4;

#if __has_builtin(__builtin_amdgcn_tensor_load_to_lds)
#define HAVE_TDM 1
#else
#define HAVE_TDM 0
#endif

#if HAVE_TDM
static __device__ inline void tdm_wait0() {
#if __has_builtin(__builtin_amdgcn_s_wait_tensorcnt)
    __builtin_amdgcn_s_wait_tensorcnt(0);
#else
    asm volatile("s_wait_tensorcnt 0x0" ::: "memory");
#endif
}

// LDS byte offset of a __shared__ object (generic -> as3 -> int)
static __device__ inline unsigned lds_offset_of(void* p) {
    return (unsigned)(unsigned long long)(__attribute__((address_space(3))) void*)p;
}

// TDM D# for a dense 2D bf16 tile: 32 (dim0, contiguous) x 128 (dim1) rows,
// row stride H_DIM elements, loaded to LDS as [128][32] (row-major append).
static __device__ inline void tdm_load_b_tile(const bf16_t* gsrc, unsigned lds_byte) {
    unsigned long long ga = (unsigned long long)(const void*)gsrc;
    u32x4 g0;
    g0[0] = 1u;                                     // count=1, user mode
    g0[1] = lds_byte;                               // lds_addr
    g0[2] = (unsigned)(ga & 0xffffffffu);           // global_addr[31:0]
    g0[3] = (unsigned)((ga >> 32) & 0x01ffffffu)    // global_addr[56:32]
          | (2u << 30);                             // type = image
    i32x8 g1;
    g1[0] = 0x00010000;                             // wg_mask=0, data_size=1 (2B)
    g1[1] = (int)((H_DIM & 0xffff) << 16);          // tensor_dim0 lo16 (=2048)
    g1[2] = (int)(((H_DIM >> 16) & 0xffff)          // tensor_dim0 hi16
          | ((F_DIM & 0xffff) << 16));              // tensor_dim1 lo16 (=2048)
    g1[3] = (int)(((F_DIM >> 16) & 0xffff)          // tensor_dim1 hi16
          | (32 << 16));                            // tile_dim0 = 32 (K, contiguous)
    g1[4] = 128;                                    // tile_dim1 = 128 rows, tile_dim2=0
    g1[5] = H_DIM;                                  // tensor_dim0_stride lo32 (=2048)
    g1[6] = 0;                                      // stride hi / dim1_stride lo
    g1[7] = 0;
    i32x4 gz4 = {0, 0, 0, 0};                       // groups 2/3 unused (2D tensor)
    i32x8 gz8 = {0, 0, 0, 0, 0, 0, 0, 0};           // 6-arg toolchain form
    __builtin_amdgcn_tensor_load_to_lds(g0, g1, gz4, gz4, gz8, 0);
}
#endif

// ---------------------------------------------------------------------------
// Kernel: expert weights fp32 [E,H,F] -> bf16 TRANSPOSED [E,F,H].
// Tiled 32x32 LDS transpose: coalesced on both sides. One-time pass so the
// GEMM's B operand (column n, K-contiguous) is contiguous in memory.
// ---------------------------------------------------------------------------
__global__ __launch_bounds__(256) void moe_convert_wT_kernel(
    const float* __restrict__ w, bf16_t* __restrict__ wT)
{
    __shared__ bf16_t tile[32][33];
    const int e  = blockIdx.z;
    const int h0 = blockIdx.y * 32;
    const int f0 = blockIdx.x * 32;
    const int tx = threadIdx.x & 31, ty = threadIdx.x >> 5;
    const float* src = w + ((long)e * H_DIM + h0) * F_DIM + f0;
    #pragma unroll
    for (int r = ty; r < 32; r += 8)
        tile[r][tx] = (bf16_t)src[(long)r * F_DIM + tx];
    __syncthreads();
    bf16_t* dst = wT + ((long)e * F_DIM + f0) * H_DIM + h0;
    #pragma unroll
    for (int r = ty; r < 32; r += 8)
        dst[(long)r * H_DIM + tx] = tile[tx][r];
}

// ---------------------------------------------------------------------------
// Kernel: fused LayerNorm + router softmax/top-2 + expert dispatch.
// One block (256 threads = 8 wave32) per token. Deterministic reductions.
// ---------------------------------------------------------------------------
__global__ __launch_bounds__(256) void moe_ln_router_kernel(
    const float* __restrict__ x,
    const float* __restrict__ gamma,
    const float* __restrict__ beta,
    const float* __restrict__ rw,   // [E,H]
    const float* __restrict__ rb,   // [E]
    bf16_t* __restrict__ xn_out,    // [N,H] bf16
    int*    __restrict__ counts,    // [E]
    int*    __restrict__ tok,       // [E,N]
    float*  __restrict__ wgt)       // [E,N]
{
    const int n    = blockIdx.x;
    const int t    = threadIdx.x;
    const int wid  = t >> 5;
    const int lane = t & 31;

    __shared__ float warr[8][N_EXP];
    __shared__ float sMu, sRstd;

    const float* xr = x + (long)n * H_DIM;
    float xv[8];
    {
        v4f a = *(const v4f*)(xr + t * 8);
        v4f b = *(const v4f*)(xr + t * 8 + 4);
        xv[0]=a[0]; xv[1]=a[1]; xv[2]=a[2]; xv[3]=a[3];
        xv[4]=b[0]; xv[5]=b[1]; xv[6]=b[2]; xv[7]=b[3];
    }

    float s = 0.f, ss = 0.f;
    #pragma unroll
    for (int j = 0; j < 8; ++j) { s += xv[j]; ss += xv[j]*xv[j]; }
    #pragma unroll
    for (int m = 16; m; m >>= 1) { s += __shfl_xor(s, m); ss += __shfl_xor(ss, m); }
    if (lane == 0) { warr[wid][0] = s; warr[wid][1] = ss; }
    __syncthreads();
    if (t == 0) {
        float S = 0.f, SS = 0.f;
        #pragma unroll
        for (int w = 0; w < 8; ++w) { S += warr[w][0]; SS += warr[w][1]; }
        float mu  = S * (1.0f / H_DIM);
        float var = SS * (1.0f / H_DIM) - mu * mu;
        sMu = mu;
        sRstd = rsqrtf(var + LN_EPS);
    }
    __syncthreads();
    const float mu = sMu, rstd = sRstd;

    float xn[8];
    #pragma unroll
    for (int j = 0; j < 8; ++j) {
        int idx = t * 8 + j;
        xn[j] = (xv[j] - mu) * rstd * gamma[idx] + beta[idx];
    }
    {
        v8bf xb;
        #pragma unroll
        for (int j = 0; j < 8; ++j) xb[j] = (bf16_t)xn[j];
        *(v8bf*)(xn_out + (long)n * H_DIM + t * 8) = xb;
    }

    float lg[N_EXP];
    #pragma unroll
    for (int e = 0; e < N_EXP; ++e) {
        const float* rwe = rw + (long)e * H_DIM + t * 8;
        v4f a = *(const v4f*)(rwe);
        v4f b = *(const v4f*)(rwe + 4);
        lg[e] = xn[0]*a[0] + xn[1]*a[1] + xn[2]*a[2] + xn[3]*a[3]
              + xn[4]*b[0] + xn[5]*b[1] + xn[6]*b[2] + xn[7]*b[3];
    }
    #pragma unroll
    for (int m = 16; m; m >>= 1) {
        #pragma unroll
        for (int e = 0; e < N_EXP; ++e) lg[e] += __shfl_xor(lg[e], m);
    }
    if (lane == 0) {
        #pragma unroll
        for (int e = 0; e < N_EXP; ++e) warr[wid][e] = lg[e];
    }
    __syncthreads();
    if (t == 0) {
        float L[N_EXP];
        #pragma unroll
        for (int e = 0; e < N_EXP; ++e) {
            float a = 0.f;
            #pragma unroll
            for (int w = 0; w < 8; ++w) a += warr[w][e];
            L[e] = a + rb[e];
        }
        int i0 = 0;
        #pragma unroll
        for (int e = 1; e < N_EXP; ++e) if (L[e] > L[i0]) i0 = e;
        int i1 = (i0 == 0) ? 1 : 0;
        #pragma unroll
        for (int e = 0; e < N_EXP; ++e) if (e != i0 && L[e] > L[i1]) i1 = e;
        float d  = __expf(L[i1] - L[i0]);
        float w0 = 1.0f / (1.0f + d);
        float w1 = d / (1.0f + d);
        int s0 = atomicAdd(&counts[i0], 1);
        tok[i0 * N_TOK + s0] = n;  wgt[i0 * N_TOK + s0] = w0;
        int s1 = atomicAdd(&counts[i1], 1);
        tok[i1 * N_TOK + s1] = n;  wgt[i1 * N_TOK + s1] = w1;
    }
}

// ---------------------------------------------------------------------------
// Kernel: per-expert gathered GEMM, 128x128 tile, K-step 32, double-buffered
// LDS. A (gathered tokens) via coalesced vector loads; B^T tile via the
// Tensor Data Mover when available. 8 waves, each 32x64 -> 8 WMMAs per step.
// Epilogue: out[token, col] += w * (acc + b_e[col]) via global f32 atomics
// (exactly TOP_K=2 commutative adds per element -> bitwise deterministic).
// ---------------------------------------------------------------------------
__global__ __launch_bounds__(256) void moe_expert_gemm_kernel(
    const bf16_t* __restrict__ xn,     // [N,H] bf16
    const bf16_t* __restrict__ wT,     // [E,F,H] bf16 (transposed weights)
    const float*  __restrict__ eb,     // [E,F]
    const int*    __restrict__ counts, // [E]
    const int*    __restrict__ tok,    // [E,N]
    const float*  __restrict__ wgt,    // [E,N]
    float* __restrict__ out)           // [N,F]
{
    const int e   = blockIdx.z;
    const int cnt = counts[e];
    const int m0  = blockIdx.y * 128;
    if (m0 >= cnt) return;
    const int n0  = blockIdx.x * 128;
    const int t   = threadIdx.x;

    __shared__ __align__(32) bf16_t Alds[2][128 * 32];  // [m][k]
    __shared__ __align__(32) bf16_t Blds[2][128 * 32];  // [n][k]
    __shared__ int   sTok[128];
    __shared__ float sWgt[128];

    if (t < 128) {
        int slot = m0 + t;
        if (slot < cnt) { sTok[t] = tok[e * N_TOK + slot]; sWgt[t] = wgt[e * N_TOK + slot]; }
        else            { sTok[t] = 0;                     sWgt[t] = 0.f; }
    }
    __syncthreads();

    const int w    = t >> 5;
    const int lane = t & 31;
    const int lo   = lane & 15;
    const int hi   = lane >> 4;
    const int mw   = (w & 3) * 32;   // wave M offset (2 subtiles of 16)
    const int nw   = (w >> 2) * 64;  // wave N offset (4 subtiles of 16)

    v8f acc[2][4];
    #pragma unroll
    for (int mi = 0; mi < 2; ++mi)
        #pragma unroll
        for (int ni = 0; ni < 4; ++ni) acc[mi][ni] = (v8f){};

    // per-thread staging indices: 128 rows x 2 threads/row, 16 elems each
    const int  arow  = t >> 1;
    const int  ahalf = (t & 1) * 16;
    const long abase = (long)sTok[arow] * H_DIM + ahalf;
#if !HAVE_TDM
    const long bbase = ((long)e * F_DIM + n0 + arow) * H_DIM + ahalf;
#endif
    const bf16_t* btile0 = wT + ((long)e * F_DIM + n0) * H_DIM;  // + k0 per step

    v8bf a0, a1;
#if !HAVE_TDM
    v8bf b0, b1;
#endif

    // ---- prologue: stage k0 = 0 into buffer 0
    a0 = *(const v8bf*)(xn + abase);
    a1 = *(const v8bf*)(xn + abase + 8);
#if HAVE_TDM
    if (t < 32) tdm_load_b_tile(btile0, lds_offset_of(&Blds[0][0]));
#else
    b0 = *(const v8bf*)(wT + bbase);
    b1 = *(const v8bf*)(wT + bbase + 8);
#endif
    *(v8bf*)(&Alds[0][arow * 32 + ahalf])     = a0;
    *(v8bf*)(&Alds[0][arow * 32 + ahalf + 8]) = a1;
#if HAVE_TDM
    if (t < 32) tdm_wait0();
#else
    *(v8bf*)(&Blds[0][arow * 32 + ahalf])     = b0;
    *(v8bf*)(&Blds[0][arow * 32 + ahalf + 8]) = b1;
#endif
    __syncthreads();

    int cur = 0;
    for (int k0 = 0; k0 < H_DIM; k0 += 32) {
        const bool more = (k0 + 32) < H_DIM;
        const int  nxt  = cur ^ 1;

        // prefetch next K-slab: A -> regs, B -> LDS[nxt] via TDM (overlaps WMMA)
        if (more) {
            a0 = *(const v8bf*)(xn + abase + k0 + 32);
            a1 = *(const v8bf*)(xn + abase + k0 + 40);
#if HAVE_TDM
            if (t < 32) tdm_load_b_tile(btile0 + k0 + 32, lds_offset_of(&Blds[nxt][0]));
#else
            b0 = *(const v8bf*)(wT + bbase + k0 + 32);
            b1 = *(const v8bf*)(wT + bbase + k0 + 40);
#endif
        }

        // compute on current buffer: 2 A frags x 4 B frags = 8 WMMAs
        #pragma unroll
        for (int mi = 0; mi < 2; ++mi) {
            const int mrow = mw + mi * 16 + lo;
            v8bf al = *(const v8bf*)(&Alds[cur][mrow * 32 + 16 * hi]);
            v8bf ah = *(const v8bf*)(&Alds[cur][mrow * 32 + 16 * hi + 8]);
            v16bf af = __builtin_shufflevector(al, ah,
                         0,1,2,3,4,5,6,7,8,9,10,11,12,13,14,15);
            #pragma unroll
            for (int ni = 0; ni < 4; ++ni) {
                v16bf bfr = *(const v16bf*)(&Blds[cur][(nw + ni * 16 + lo) * 32 + 16 * hi]);
                acc[mi][ni] = __builtin_amdgcn_wmma_f32_16x16x32_bf16(
                                  false, af, false, bfr, (short)0, acc[mi][ni],
                                  false, false);
            }
        }

        if (more) {
            *(v8bf*)(&Alds[nxt][arow * 32 + ahalf])     = a0;
            *(v8bf*)(&Alds[nxt][arow * 32 + ahalf + 8]) = a1;
#if HAVE_TDM
            if (t < 32) tdm_wait0();
#else
            *(v8bf*)(&Blds[nxt][arow * 32 + ahalf])     = b0;
            *(v8bf*)(&Blds[nxt][arow * 32 + ahalf + 8]) = b1;
#endif
        }
        __syncthreads();
        cur ^= 1;
    }

    // ---- epilogue: scatter-add w * (acc + bias)
    #pragma unroll
    for (int mi = 0; mi < 2; ++mi) {
        #pragma unroll
        for (int ni = 0; ni < 4; ++ni) {
            const int col = n0 + nw + ni * 16 + lo;
            const float be = eb[(long)e * F_DIM + col];
            #pragma unroll
            for (int v = 0; v < 8; ++v) {
                const int ml = mw + mi * 16 + v + 8 * hi;
                if (m0 + ml < cnt) {
                    atomicAdd(out + (long)sTok[ml] * F_DIM + col,
                              sWgt[ml] * (acc[mi][ni][v] + be));
                }
            }
        }
    }
}

// ---------------------------------------------------------------------------
// Launch
// ---------------------------------------------------------------------------
extern "C" void kernel_launch(void* const* d_in, const int* in_sizes, int n_in,
                              void* d_out, int out_size, void* d_ws, size_t ws_size,
                              hipStream_t stream) {
    const float* x        = (const float*)d_in[0];
    const float* ln_gamma = (const float*)d_in[1];
    const float* ln_beta  = (const float*)d_in[2];
    const float* router_w = (const float*)d_in[3];
    const float* router_b = (const float*)d_in[4];
    const float* expert_w = (const float*)d_in[5];
    const float* expert_b = (const float*)d_in[6];
    float* out = (float*)d_out;

    // workspace layout
    char* ws = (char*)d_ws;
    bf16_t* xn_bf  = (bf16_t*)(ws);                                  // N*H*2
    bf16_t* wT_bf  = (bf16_t*)(ws + (size_t)N_TOK * H_DIM * 2);      // E*F*H*2
    char*   p      = ws + (size_t)N_TOK * H_DIM * 2 + (size_t)N_EXP * H_DIM * F_DIM * 2;
    int*    tok    = (int*)(p);
    float*  wgt    = (float*)(p + (size_t)N_EXP * N_TOK * 4);
    int*    counts = (int*)(p + (size_t)N_EXP * N_TOK * 8);

    (void)hipMemsetAsync(out,    0, (size_t)N_TOK * F_DIM * sizeof(float), stream);
    (void)hipMemsetAsync(counts, 0, (size_t)N_EXP * sizeof(int), stream);

    // 1) weights fp32 [E,H,F] -> bf16 transposed [E,F,H]
    {
        dim3 grid(F_DIM / 32, H_DIM / 32, N_EXP);
        moe_convert_wT_kernel<<<grid, 256, 0, stream>>>(expert_w, wT_bf);
    }

    // 2) fused LayerNorm + router + dispatch
    moe_ln_router_kernel<<<N_TOK, 256, 0, stream>>>(
        x, ln_gamma, ln_beta, router_w, router_b, xn_bf, counts, tok, wgt);

    // 3) per-expert gathered bf16 WMMA GEMM + scatter
    {
        dim3 grid(F_DIM / 128, N_TOK / 128, N_EXP);
        moe_expert_gemm_kernel<<<grid, 256, 0, stream>>>(
            xn_bf, wT_bf, expert_b, counts, tok, wgt, out);
    }
}
